// LocalFeatureAggregation_58394375356476
// MI455X (gfx1250) — compile-verified, matrix operand order checked by reading, not run on hardware
//
#include <hip/hip_runtime.h>
#include <hip/hip_bf16.h>

// ---------------- constants ----------------
constexpr int BATCH = 4;
constexpr int CIN   = 8;
constexpr int CH    = 32;      // feature channels
constexpr int NPTS  = 32768;   // points per batch
constexpr int NCAND = 64;
constexpr int KNN   = 16;
constexpr int P     = BATCH * NPTS;   // 131072 total points

// BN stages: 0=pre 1=sc 2=a1.bn2 3=a1.bnm 4=a2.bn2 5=a2.bnm 6=post
// stats[stage*64 + c]      : channel sum
// stats[stage*64 + 32 + c] : channel sumsq
// stats[(7+stage)*64 + c]  : BN scale ; +32+c : BN shift
#define DEV __device__ __forceinline__

typedef __attribute__((ext_vector_type(16))) __bf16 v16bf;
typedef __attribute__((ext_vector_type(8)))  float  v8f;
typedef __attribute__((ext_vector_type(4)))  unsigned v4u;
typedef __attribute__((ext_vector_type(8)))  unsigned v8u;

// ---------------- helpers ----------------
DEV unsigned short f2bfu(float f) {
  unsigned u = __builtin_bit_cast(unsigned, f);
  unsigned r = u + 0x7FFFu + ((u >> 16) & 1u);
  return (unsigned short)(r >> 16);
}
DEV __bf16 f2bf(float f) {
  unsigned short h = f2bfu(f);
  return __builtin_bit_cast(__bf16, h);
}
DEV float bfu2f(unsigned short h) {
  unsigned u = ((unsigned)h) << 16;
  return __builtin_bit_cast(float, u);
}
DEV unsigned lds_off(const void* p) {
  // generic LDS pointer: addr[31:0] is the LDS byte offset (ISA 10.2 aperture map)
  return (unsigned)(unsigned long long)p;
}

// A-matrix 16x32 bf16 fragment (rows t*16..t*16+15) from f32 row-major (32,32) weights.
// Layout (ISA 7.12.2): lane l holds row m=l&15; VGPR j holds K = 16*(j/4)+8*(l/16)+2*(j%4)+{0,1}.
DEV v16bf load_w_frag(const float* __restrict__ w, int t, int lane) {
  int m = (lane & 15) + t * 16;
  int half = lane >> 4;
  v16bf a;
#pragma unroll
  for (int j = 0; j < 8; ++j) {
    int k0 = ((j >> 2) << 4) + (half << 3) + ((j & 3) << 1);
    a[2 * j]     = f2bf(w[m * CH + k0]);
    a[2 * j + 1] = f2bf(w[m * CH + k0 + 1]);
  }
  return a;
}

// B-matrix 32x16 bf16 fragment from an LDS tile of 16 positions x 32 bf16 (64B rows).
DEV v16bf frag_from_lds_bf16(const unsigned int* tile_u32, int lane) {
  int n = lane & 15, half = lane >> 4;
  v16bf b;
#pragma unroll
  for (int j = 0; j < 8; ++j) {
    int k0 = ((j >> 2) << 4) + (half << 3) + ((j & 3) << 1);
    unsigned int wu = tile_u32[n * 16 + (k0 >> 1)];
    b[2 * j]     = __builtin_bit_cast(__bf16, (unsigned short)(wu & 0xFFFFu));
    b[2 * j + 1] = __builtin_bit_cast(__bf16, (unsigned short)(wu >> 16));
  }
  return b;
}

// B-matrix 32x16 bf16 fragment from LDS f32 tile (16 positions x 32 f32).
DEV v16bf frag_from_lds_f32(const float* pool, int lane) {
  int n = lane & 15, half = lane >> 4;
  v16bf b;
#pragma unroll
  for (int j = 0; j < 8; ++j) {
    int k0 = ((j >> 2) << 4) + (half << 3) + ((j & 3) << 1);
    b[2 * j]     = f2bf(pool[n * CH + k0]);
    b[2 * j + 1] = f2bf(pool[n * CH + k0 + 1]);
  }
  return b;
}

// Async gather of 16 neighbor rows (64B bf16 each) straight into an LDS tile:
// 2 x global_load_async_to_lds_b128 per lane, tracked with ASYNCcnt.
DEV void gather_tile_async(const unsigned short* __restrict__ hb,
                           const int* __restrict__ gidx_p,
                           unsigned int* tile, int lane) {
  unsigned base = lds_off(tile);
#pragma unroll
  for (int r = 0; r < 2; ++r) {
    int chunk = r * 32 + lane;      // 64 16-byte chunks
    int nbr = chunk >> 2, q = chunk & 3;
    int g = gidx_p[nbr];
    unsigned long long ga =
        (unsigned long long)(const void*)(hb + (size_t)g * CH) + (unsigned)(q * 16);
    unsigned lds = base + (unsigned)(nbr * 64 + q * 16);
    asm volatile("global_load_async_to_lds_b128 %0, %1, off"
                 :: "v"(lds), "v"(ga) : "memory");
  }
}

DEV void wait_async0() { asm volatile("s_wait_asynccnt 0x0" ::: "memory"); }
DEV void wait_async2() { asm volatile("s_wait_asynccnt 0x2" ::: "memory"); }
DEV void lds_fence()   { asm volatile("s_wait_dscnt 0x0"    ::: "memory"); }

DEV unsigned rfl(unsigned x) { return (unsigned)__builtin_amdgcn_readfirstlane((int)x); }

// Tensor Data Mover: DMA one contiguous 1KB tile (16 points x 64B bf16) into LDS.
// D# group0/group1 built in SGPRs; VADDR2/VADDR3 omitted (NULL) => <=2D tensor.
DEV void tdm_load_tile(const unsigned short* __restrict__ hb, int p0,
                       unsigned int* tile) {
  unsigned long long ga = (unsigned long long)(const void*)(hb + (size_t)p0 * CH);
  unsigned ga_lo = rfl((unsigned)ga);
  unsigned ga_hi = rfl((unsigned)(ga >> 32));
  unsigned lds   = rfl(lds_off(tile));
  v4u g0;
  g0[0] = 1u;                                   // count=1, is_restore=0, gather off
  g0[1] = lds;                                  // lds_addr
  g0[2] = ga_lo;                                // global_addr[31:0]
  g0[3] = (ga_hi & 0x01FFFFFFu) | (2u << 30);   // global_addr[56:32] | type=2
  v8u g1;
  g1[0] = 3u << 16;       // workgroup_mask=0, data_size=3 (8B)
  g1[1] = 128u << 16;     // tensor_dim0 = 128 (8B units)  -> bits[79:48]
  g1[2] = 1u << 16;       // tensor_dim1 = 1               -> bits[111:80]
  g1[3] = 128u << 16;     // tile_dim0 = 128               -> bits[127:112]
  g1[4] = 1u;             // tile_dim1 = 1, tile_dim2 = 0
  g1[5] = 128u;           // tensor_dim0_stride = 128
  g1[6] = 0u;
  g1[7] = 0u;
  asm volatile("tensor_load_to_lds %0, %1" :: "s"(g0), "s"(g1) : "memory");
}

// ---------------- kernels ----------------
__global__ void k_zero(float* stats) {
  int i = blockIdx.x * blockDim.x + threadIdx.x;
  if (i < 14 * 64) stats[i] = 0.f;
}

__global__ __launch_bounds__(256)
void k_pre_sc(const float* __restrict__ x, const float* __restrict__ wpre,
              const float* __restrict__ wsc, float* __restrict__ lin_pre,
              float* __restrict__ lin_sc, float* stats) {
  __shared__ float acc[4][CH];
  int t = threadIdx.x;
  if (t < 128) ((float*)acc)[t] = 0.f;
  __syncthreads();
  int p = blockIdx.x * blockDim.x + t;
  int b = p >> 15, n = p & (NPTS - 1);
  float xv[CIN];
#pragma unroll
  for (int c = 0; c < CIN; ++c) xv[c] = x[(b * CIN + c) * NPTS + n];
#pragma unroll
  for (int oc = 0; oc < CH; ++oc) {
    float a = 0.f, s = 0.f;
#pragma unroll
    for (int c = 0; c < CIN; ++c) {
      a = fmaf(wpre[oc * CIN + c], xv[c], a);
      s = fmaf(wsc[oc * CIN + c], xv[c], s);
    }
    lin_pre[(size_t)p * CH + oc] = a;
    lin_sc[(size_t)p * CH + oc] = s;
    atomicAdd(&acc[0][oc], a); atomicAdd(&acc[1][oc], a * a);
    atomicAdd(&acc[2][oc], s); atomicAdd(&acc[3][oc], s * s);
  }
  __syncthreads();
  if (t < CH) {
    atomicAdd(&stats[0 * 64 + t],      acc[0][t]);
    atomicAdd(&stats[0 * 64 + 32 + t], acc[1][t]);
    atomicAdd(&stats[1 * 64 + t],      acc[2][t]);
    atomicAdd(&stats[1 * 64 + 32 + t], acc[3][t]);
  }
}

__global__ void k_finalize(float* stats, const float* __restrict__ g,
                           const float* __restrict__ bb, int stage, float invcnt) {
  int c = threadIdx.x;
  if (c >= CH) return;
  float s = stats[stage * 64 + c], q = stats[stage * 64 + 32 + c];
  float m = s * invcnt;
  float v = q * invcnt - m * m;
  float sc = g[c] * rsqrtf(v + 1e-5f);
  stats[(7 + stage) * 64 + c] = sc;
  stats[(7 + stage) * 64 + 32 + c] = bb[c] - m * sc;
}

__global__ __launch_bounds__(256)
void k_apply(const float* __restrict__ lin, const float* __restrict__ stats,
             int stage, float* __restrict__ hf, unsigned short* __restrict__ hb) {
  size_t i = (size_t)blockIdx.x * blockDim.x + threadIdx.x;  // over P*CH
  int c = (int)(i & 31);
  float v = fmaf(stats[(7 + stage) * 64 + c], lin[i], stats[(7 + stage) * 64 + 32 + c]);
  v = v > 0.f ? v : 0.f;
  hf[i] = v;
  hb[i] = f2bfu(v);
}

__global__ __launch_bounds__(256)
void k_knn(const float* __restrict__ hf, const int* __restrict__ cand,
           int* __restrict__ gidx) {
  int p = blockIdx.x * blockDim.x + threadIdx.x;
  int b = p >> 15;
  int base = b * NPTS;
  float own[CH];
  const float* op = hf + (size_t)p * CH;
#pragma unroll
  for (int c = 0; c < CH; ++c) own[c] = op[c];
  float bd[KNN]; int bi[KNN];
#pragma unroll
  for (int k = 0; k < KNN; ++k) { bd[k] = 3.4e38f; bi[k] = base; }
  const int* cp = cand + (size_t)p * NCAND;
  for (int j = 0; j < NCAND; ++j) {
    int ci = cp[j];
    const float* qp = hf + (size_t)(base + ci) * CH;
    if (j + 2 < NCAND)
      __builtin_prefetch(hf + (size_t)(base + cp[j + 2]) * CH, 0, 1);
    float d = 0.f;
#pragma unroll
    for (int c = 0; c < CH; ++c) { float df = own[c] - qp[c]; d = fmaf(df, df, d); }
    if (d < bd[KNN - 1]) {
      int k = KNN - 1;
      while (k > 0 && bd[k - 1] > d) { bd[k] = bd[k - 1]; bi[k] = bi[k - 1]; --k; }
      bd[k] = d; bi[k] = base + ci;
    }
  }
#pragma unroll
  for (int k = 0; k < KNN; ++k) gidx[(size_t)p * KNN + k] = bi[k];
}

// Stats pass over lin1 = w1 . gathered neighbors: async double-buffered gather,
// 2 WMMAs per 16-position tile.
__global__ __launch_bounds__(256)
void k_att_stats(const unsigned short* __restrict__ hb, const int* __restrict__ gidx,
                 const float* __restrict__ w1, float* stats, int stage) {
  __shared__ __align__(16) unsigned int tiles[8][2][256];
  __shared__ float ssum[CH], ssq[CH];
  int lane = threadIdx.x & 31, wid = threadIdx.x >> 5;
  if (threadIdx.x < CH) { ssum[threadIdx.x] = 0.f; ssq[threadIdx.x] = 0.f; }
  __syncthreads();
  v16bf a0 = load_w_frag(w1, 0, lane);
  v16bf a1 = load_w_frag(w1, 1, lane);
  float as[2][8] = {}, aq[2][8] = {};
  int wave = blockIdx.x * 8 + wid;
  int nwaves = gridDim.x * 8;
  int cur = 0;
  if (wave < P) gather_tile_async(hb, gidx + (size_t)wave * KNN, tiles[wid][0], lane);
  for (int p = wave; p < P; p += nwaves) {
    int pn = p + nwaves;
    bool pre = pn < P;
    if (pre) gather_tile_async(hb, gidx + (size_t)pn * KNN, tiles[wid][cur ^ 1], lane);
    if (pre) wait_async2(); else wait_async0();
    v16bf bfr = frag_from_lds_bf16(tiles[wid][cur], lane);
    v8f d0 = {}, d1 = {};
    d0 = __builtin_amdgcn_wmma_f32_16x16x32_bf16(false, a0, false, bfr, (short)0, d0, false, false);
    d1 = __builtin_amdgcn_wmma_f32_16x16x32_bf16(false, a1, false, bfr, (short)0, d1, false, false);
#pragma unroll
    for (int r = 0; r < 8; ++r) {
      as[0][r] += d0[r]; aq[0][r] += d0[r] * d0[r];
      as[1][r] += d1[r]; aq[1][r] += d1[r] * d1[r];
    }
    cur ^= 1;
  }
  int half = lane >> 4;
#pragma unroll
  for (int t = 0; t < 2; ++t)
#pragma unroll
    for (int r = 0; r < 8; ++r) {
      int ch = t * 16 + half * 8 + r;
      atomicAdd(&ssum[ch], as[t][r]);
      atomicAdd(&ssq[ch], aq[t][r]);
    }
  __syncthreads();
  if (threadIdx.x < CH) {
    atomicAdd(&stats[stage * 64 + threadIdx.x], ssum[threadIdx.x]);
    atomicAdd(&stats[stage * 64 + 32 + threadIdx.x], ssq[threadIdx.x]);
  }
}

// Pool pass: recompute lin1 (async double-buffered gather + WMMA), BN+ReLU,
// score=w2.s+b2, in-wave softmax over K, weighted pool, wm WMMA + bnm stats.
__global__ __launch_bounds__(256)
void k_att_pool(const unsigned short* __restrict__ hb, const int* __restrict__ gidx,
                const float* __restrict__ w1, const float* __restrict__ w2,
                const float* __restrict__ b2, const float* __restrict__ wm,
                float* stats, int stBN2, int stBNM, float* __restrict__ lin_out) {
  __shared__ __align__(16) unsigned int tiles[8][2][256];
  __shared__ float pool[8][16 * CH];
  __shared__ float wls[8][16];
  __shared__ float ssum[CH], ssq[CH];
  int lane = threadIdx.x & 31, wid = threadIdx.x >> 5;
  int half = lane >> 4, col = lane & 15;
  if (threadIdx.x < CH) { ssum[threadIdx.x] = 0.f; ssq[threadIdx.x] = 0.f; }
  __syncthreads();
  v16bf a0 = load_w_frag(w1, 0, lane);
  v16bf a1 = load_w_frag(w1, 1, lane);
  v16bf m0 = load_w_frag(wm, 0, lane);
  v16bf m1 = load_w_frag(wm, 1, lane);
  float bnsc[2][8], bnsh[2][8], w2v[2][8];
#pragma unroll
  for (int t = 0; t < 2; ++t)
#pragma unroll
    for (int r = 0; r < 8; ++r) {
      int ch = t * 16 + half * 8 + r;
      bnsc[t][r] = stats[(7 + stBN2) * 64 + ch];
      bnsh[t][r] = stats[(7 + stBN2) * 64 + 32 + ch];
      w2v[t][r]  = w2[ch];
    }
  float b2v = b2[0];
  float as[2][8] = {}, aq[2][8] = {};
  int wave = blockIdx.x * 8 + wid;
  int nwaves = gridDim.x * 8;
  for (int mt = wave; mt < P / 16; mt += nwaves) {
    int p0 = mt * 16;
    int cur = 0;
    gather_tile_async(hb, gidx + (size_t)p0 * KNN, tiles[wid][0], lane);
    for (int pt = 0; pt < 16; ++pt) {
      bool pre = pt < 15;
      if (pre)
        gather_tile_async(hb, gidx + (size_t)(p0 + pt + 1) * KNN,
                          tiles[wid][cur ^ 1], lane);
      if (pre) wait_async2(); else wait_async0();
      const unsigned int* tile = tiles[wid][cur];
      v16bf bfr = frag_from_lds_bf16(tile, lane);
      v8f d0 = {}, d1 = {};
      d0 = __builtin_amdgcn_wmma_f32_16x16x32_bf16(false, a0, false, bfr, (short)0, d0, false, false);
      d1 = __builtin_amdgcn_wmma_f32_16x16x32_bf16(false, a1, false, bfr, (short)0, d1, false, false);
      float part = 0.f;
#pragma unroll
      for (int r = 0; r < 8; ++r) {
        float s0 = fmaf(bnsc[0][r], d0[r], bnsh[0][r]); s0 = s0 > 0.f ? s0 : 0.f;
        float s1 = fmaf(bnsc[1][r], d1[r], bnsh[1][r]); s1 = s1 > 0.f ? s1 : 0.f;
        part = fmaf(w2v[0][r], s0, part);
        part = fmaf(w2v[1][r], s1, part);
      }
      float score = part + __shfl_xor(part, 16, 32) + b2v;
      float mx = score;
#pragma unroll
      for (int s = 8; s >= 1; s >>= 1) mx = fmaxf(mx, __shfl_xor(mx, s, 32));
      float e = __expf(score - mx);
      float se = e;
#pragma unroll
      for (int s = 8; s >= 1; s >>= 1) se += __shfl_xor(se, s, 32);
      float wk = e / se;
      if (half == 0) wls[wid][col] = wk;
      lds_fence();
      // pooled value for channel c = lane from the resident tile
      float pc = 0.f;
#pragma unroll
      for (int k = 0; k < KNN; ++k) {
        unsigned int wu = tile[k * 16 + (lane >> 1)];
        unsigned short hbits = (lane & 1) ? (unsigned short)(wu >> 16)
                                          : (unsigned short)(wu & 0xFFFFu);
        pc = fmaf(wls[wid][k], bfu2f(hbits), pc);
      }
      pool[wid][pt * CH + lane] = pc;
      cur ^= 1;
    }
    lds_fence();
    v16bf bp = frag_from_lds_f32(pool[wid], lane);
    v8f e0 = {}, e1 = {};
    e0 = __builtin_amdgcn_wmma_f32_16x16x32_bf16(false, m0, false, bp, (short)0, e0, false, false);
    e1 = __builtin_amdgcn_wmma_f32_16x16x32_bf16(false, m1, false, bp, (short)0, e1, false, false);
    int pg = p0 + col;
#pragma unroll
    for (int r = 0; r < 8; ++r) {
      int ch0 = half * 8 + r, ch1 = 16 + half * 8 + r;
      float v0 = e0[r], v1 = e1[r];
      lin_out[(size_t)pg * CH + ch0] = v0;
      lin_out[(size_t)pg * CH + ch1] = v1;
      as[0][r] += v0; aq[0][r] += v0 * v0;
      as[1][r] += v1; aq[1][r] += v1 * v1;
    }
  }
#pragma unroll
  for (int t = 0; t < 2; ++t)
#pragma unroll
    for (int r = 0; r < 8; ++r) {
      int ch = t * 16 + half * 8 + r;
      atomicAdd(&ssum[ch], as[t][r]);
      atomicAdd(&ssq[ch], aq[t][r]);
    }
  __syncthreads();
  if (threadIdx.x < CH) {
    atomicAdd(&stats[stBNM * 64 + threadIdx.x], ssum[threadIdx.x]);
    atomicAdd(&stats[stBNM * 64 + 32 + threadIdx.x], ssq[threadIdx.x]);
  }
}

// Dense 32x32 linear over 16-point tiles (w_post): TDM tensor_load_to_lds DMA
// stages each contiguous 1KB tile, then 2 WMMAs + BN stats.
__global__ __launch_bounds__(256)
void k_lin32_stats(const unsigned short* __restrict__ hb, const float* __restrict__ w,
                   float* stats, int stage, float* __restrict__ lin_out) {
  __shared__ __align__(16) unsigned int tiles[8][256];
  __shared__ float ssum[CH], ssq[CH];
  int lane = threadIdx.x & 31, wid = threadIdx.x >> 5;
  int half = lane >> 4, col = lane & 15;
  if (threadIdx.x < CH) { ssum[threadIdx.x] = 0.f; ssq[threadIdx.x] = 0.f; }
  __syncthreads();
  v16bf a0 = load_w_frag(w, 0, lane);
  v16bf a1 = load_w_frag(w, 1, lane);
  float as[2][8] = {}, aq[2][8] = {};
  int wave = blockIdx.x * 8 + wid;
  int nwaves = gridDim.x * 8;
  for (int mt = wave; mt < P / 16; mt += nwaves) {
    int p0 = mt * 16;
    tdm_load_tile(hb, p0, tiles[wid]);
    __builtin_amdgcn_s_wait_tensorcnt(0);
    v16bf bfr = frag_from_lds_bf16(tiles[wid], lane);
    v8f d0 = {}, d1 = {};
    d0 = __builtin_amdgcn_wmma_f32_16x16x32_bf16(false, a0, false, bfr, (short)0, d0, false, false);
    d1 = __builtin_amdgcn_wmma_f32_16x16x32_bf16(false, a1, false, bfr, (short)0, d1, false, false);
    int pg = p0 + col;
#pragma unroll
    for (int r = 0; r < 8; ++r) {
      int ch0 = half * 8 + r, ch1 = 16 + half * 8 + r;
      float v0 = d0[r], v1 = d1[r];
      lin_out[(size_t)pg * CH + ch0] = v0;
      lin_out[(size_t)pg * CH + ch1] = v1;
      as[0][r] += v0; aq[0][r] += v0 * v0;
      as[1][r] += v1; aq[1][r] += v1 * v1;
    }
  }
#pragma unroll
  for (int t = 0; t < 2; ++t)
#pragma unroll
    for (int r = 0; r < 8; ++r) {
      int ch = t * 16 + half * 8 + r;
      atomicAdd(&ssum[ch], as[t][r]);
      atomicAdd(&ssq[ch], aq[t][r]);
    }
  __syncthreads();
  if (threadIdx.x < CH) {
    atomicAdd(&stats[stage * 64 + threadIdx.x], ssum[threadIdx.x]);
    atomicAdd(&stats[stage * 64 + 32 + threadIdx.x], ssq[threadIdx.x]);
  }
}

__global__ __launch_bounds__(256)
void k_final(const float* __restrict__ linp, const float* __restrict__ sc_lin,
             const float* __restrict__ stats, float* __restrict__ out) {
  size_t i = (size_t)blockIdx.x * blockDim.x + threadIdx.x;  // P*CH
  int c = (int)(i & 31);
  int p = (int)(i >> 5);
  int b = p >> 15, n = p & (NPTS - 1);
  float hp = fmaf(stats[(7 + 6) * 64 + c], linp[i], stats[(7 + 6) * 64 + 32 + c]);
  hp = hp > 0.f ? hp : 0.f;
  float ss = fmaf(stats[(7 + 1) * 64 + c], sc_lin[i], stats[(7 + 1) * 64 + 32 + c]);
  ss = ss > 0.f ? ss : 0.f;
  float v = hp + ss;
  out[((size_t)b * CH + c) * NPTS + n] = v > 0.f ? v : 0.2f * v;
}

// ---------------- host launcher ----------------
extern "C" void kernel_launch(void* const* d_in, const int* in_sizes, int n_in,
                              void* d_out, int out_size, void* d_ws, size_t ws_size,
                              hipStream_t stream) {
  const float* x      = (const float*)d_in[0];
  const int*   cand   = (const int*)  d_in[1];
  const float* w_pre  = (const float*)d_in[2];
  const float* g_pre  = (const float*)d_in[3];
  const float* b_pre  = (const float*)d_in[4];
  const float* w_sc   = (const float*)d_in[5];
  const float* g_sc   = (const float*)d_in[6];
  const float* b_sc   = (const float*)d_in[7];
  const float* a1_w1  = (const float*)d_in[8];
  const float* a1_g1  = (const float*)d_in[9];
  const float* a1_b1  = (const float*)d_in[10];
  const float* a1_w2  = (const float*)d_in[11];
  const float* a1_b2  = (const float*)d_in[12];
  const float* a1_wm  = (const float*)d_in[13];
  const float* a1_gm  = (const float*)d_in[14];
  const float* a1_bm  = (const float*)d_in[15];
  const float* a2_w1  = (const float*)d_in[16];
  const float* a2_g1  = (const float*)d_in[17];
  const float* a2_b1  = (const float*)d_in[18];
  const float* a2_w2  = (const float*)d_in[19];
  const float* a2_b2  = (const float*)d_in[20];
  const float* a2_wm  = (const float*)d_in[21];
  const float* a2_gm  = (const float*)d_in[22];
  const float* a2_bm  = (const float*)d_in[23];
  const float* w_post = (const float*)d_in[24];
  const float* g_post = (const float*)d_in[25];
  const float* b_post = (const float*)d_in[26];

  char* ws = (char*)d_ws;
  float*          stats  = (float*)ws;           ws += 4096;
  float*          sc_lin = (float*)ws;           ws += (size_t)P * CH * 4;
  float*          hf     = (float*)ws;           ws += (size_t)P * CH * 4;
  unsigned short* hb     = (unsigned short*)ws;  ws += (size_t)P * CH * 2;
  int*            gidx   = (int*)ws;             ws += (size_t)P * KNN * 4;
  float*          lin    = (float*)ws;           ws += (size_t)P * CH * 4;

  const float invP  = 1.f / (float)P;
  const float invPK = 1.f / ((float)P * (float)KNN);

  k_zero<<<4, 256, 0, stream>>>(stats);
  // pre + shortcut linears + BN stats
  k_pre_sc<<<P / 256, 256, 0, stream>>>(x, w_pre, w_sc, lin, sc_lin, stats);
  k_finalize<<<1, 32, 0, stream>>>(stats, g_pre, b_pre, 0, invP);
  k_finalize<<<1, 32, 0, stream>>>(stats, g_sc, b_sc, 1, invP);
  k_apply<<<P * CH / 256, 256, 0, stream>>>(lin, stats, 0, hf, hb);
  // KNN over 64 candidates
  k_knn<<<P / 256, 256, 0, stream>>>(hf, cand, gidx);
  // attention block 1
  k_att_stats<<<256, 256, 0, stream>>>(hb, gidx, a1_w1, stats, 2);
  k_finalize<<<1, 32, 0, stream>>>(stats, a1_g1, a1_b1, 2, invPK);
  k_att_pool<<<256, 256, 0, stream>>>(hb, gidx, a1_w1, a1_w2, a1_b2, a1_wm,
                                      stats, 2, 3, lin);
  k_finalize<<<1, 32, 0, stream>>>(stats, a1_gm, a1_bm, 3, invP);
  k_apply<<<P * CH / 256, 256, 0, stream>>>(lin, stats, 3, hf, hb);
  // attention block 2 (same idx, new features)
  k_att_stats<<<256, 256, 0, stream>>>(hb, gidx, a2_w1, stats, 4);
  k_finalize<<<1, 32, 0, stream>>>(stats, a2_g1, a2_b1, 4, invPK);
  k_att_pool<<<256, 256, 0, stream>>>(hb, gidx, a2_w1, a2_w2, a2_b2, a2_wm,
                                      stats, 4, 5, lin);
  k_finalize<<<1, 32, 0, stream>>>(stats, a2_gm, a2_bm, 5, invP);
  k_apply<<<P * CH / 256, 256, 0, stream>>>(lin, stats, 5, hf, hb);
  // post linear + BN + ReLU + shortcut + leaky ReLU
  k_lin32_stats<<<256, 256, 0, stream>>>(hb, w_post, stats, 6, lin);
  k_finalize<<<1, 32, 0, stream>>>(stats, g_post, b_post, 6, invP);
  k_final<<<P * CH / 256, 256, 0, stream>>>(lin, sc_lin, stats, (float*)d_out);
}
